// DisentangleRNNDecoder_45268955300292
// MI455X (gfx1250) — matrix operand order repaired and checked by reading, not compiled, not run on hardware
//
#include <hip/hip_runtime.h>
#include <stdint.h>

// Problem constants (match reference)
#define VV 32000
#define BB 64
#define TT 256
#define DD 512
#define HH 1024
#define H3 (3 * HH)

typedef __attribute__((ext_vector_type(16))) __bf16 v16bf;
typedef __attribute__((ext_vector_type(8)))  float  v8f;

// ---------- scalar helpers ----------
__device__ __forceinline__ uint16_t f2bf(float f) {
    union { float f; uint32_t u; } c; c.f = f;
    uint32_t x = c.u;
    return (uint16_t)((x + 0x7FFFu + ((x >> 16) & 1u)) >> 16);  // RNE
}
__device__ __forceinline__ float bf2f(uint16_t h) {
    union { uint32_t u; float f; } c; c.u = ((uint32_t)h) << 16;
    return c.f;
}
__device__ __forceinline__ float sigmoidf_(float x) {
    return 1.0f / (1.0f + __expf(-x));
}

// ---------- WMMA fragment loaders (bf16, 16x16x32) ----------
// A (16x32, 16-bit): lanes 0-15: M=lane, K = {0..7, 16..23}; lanes 16-31: M=lane-16, K = {8..15, 24..31}
// rowp points at the start of this lane's A row (base + row*stride).
__device__ __forceinline__ v16bf load_a_frag(const uint16_t* rowp, int k) {
    const int lane = threadIdx.x & 31;
    const int koff = (lane >= 16) ? 8 : 0;
    union { uint4 q[2]; v16bf v; } f;
    f.q[0] = *reinterpret_cast<const uint4*>(rowp + k + koff);
    f.q[1] = *reinterpret_cast<const uint4*>(rowp + k + 16 + koff);
    return f.v;
}
// B (32x16, 16-bit): lane = N col (lanes 0-15 K={0..7,16..23}, lanes 16-31 K={8..15,24..31});
// colp points at the start of this lane's (transposed) weight row: WT + n*K.
__device__ __forceinline__ v16bf load_b_frag(const uint16_t* colp, int k) {
    const int lane = threadIdx.x & 31;
    const int kb = (lane >= 16) ? 8 : 0;
    union { uint4 q[2]; v16bf v; } f;
    f.q[0] = *reinterpret_cast<const uint4*>(colp + k + kb);
    f.q[1] = *reinterpret_cast<const uint4*>(colp + k + kb + 16);
    return f.v;
}
__device__ __forceinline__ v8f wmma_bf16(v16bf a, v16bf b, v8f c) {
    return __builtin_amdgcn_wmma_f32_16x16x32_bf16(false, a, false, b, (short)0, c, false, false);
}

// ---------- prep kernels ----------
// LDS-tiled transpose+convert: out[n*K + k] = bf16(in[k*N + n]).
// 32x32 tile, padded LDS to avoid bank conflicts; both global sides coalesced.
// K, N are multiples of 32 for all our weights (512/1024/3072) -> no bounds checks.
__global__ __launch_bounds__(256)
void transpose_cvt_kernel(const float* __restrict__ in,
                          uint16_t* __restrict__ out, int K, int N) {
    __shared__ uint16_t tile[32][33];
    const int n0 = blockIdx.x * 32;
    const int k0 = blockIdx.y * 32;
    const int tx = threadIdx.x;   // 0..31
    const int ty = threadIdx.y;   // 0..7
#pragma unroll
    for (int r = ty; r < 32; r += 8)
        tile[r][tx] = f2bf(in[(size_t)(k0 + r) * N + (n0 + tx)]);
    __syncthreads();
#pragma unroll
    for (int r = ty; r < 32; r += 8)
        out[(size_t)(n0 + r) * K + (k0 + tx)] = tile[tx][r];
}

// X[(b*T+t)*D + d] = bf16(emb_table[y[b*T+t]*D + d])
__global__ void gather_cvt_kernel(const int* __restrict__ y,
                                  const float* __restrict__ emb,
                                  uint16_t* __restrict__ X) {
    size_t idx = (size_t)blockIdx.x * blockDim.x + threadIdx.x;
    if (idx >= (size_t)BB * TT * DD) return;
    size_t row = idx / DD, d = idx % DD;
    int tok = y[row];
    X[idx] = f2bf(emb[(size_t)tok * DD + d]);
}

__global__ void cvt_f32_bf16_kernel(const float* __restrict__ in,
                                    uint16_t* __restrict__ out, int n) {
    int idx = blockIdx.x * blockDim.x + threadIdx.x;
    if (idx < n) out[idx] = f2bf(in[idx]);
}

// ---------- fused GRU step ----------
// grid = 64 H-col tiles; block = 128 threads = 4 waves, one wave per 16-row
// batch tile. All 4 waves consume the SAME weight-column fragments, so the
// redundant B loads hit the WGP cache instead of multiplying L2 traffic.
// Per 16x16 tile of the [B=64, H=1024] state:
//   r = sigmoid(x@Wx_r + h@Wh_r + bx_r + bh_r)    (shared accumulator)
//   z = sigmoid(x@Wx_z + h@Wh_z + bx_z + bh_z)    (shared accumulator)
//   n = tanh(x@Wx_n + bx_n + r*(h@Wh_n + bh_n))   (split accumulators)
//   h' = (1-z)*n + z*h
__global__ __launch_bounds__(128)
void gru_step_kernel(const uint16_t* __restrict__ X,     // [B*T, D] bf16
                     const uint16_t* __restrict__ WxT,   // [3H, D]  bf16 (transposed)
                     const uint16_t* __restrict__ WhT,   // [3H, H]  bf16 (transposed)
                     const float* __restrict__ bx,       // [3H]
                     const float* __restrict__ bh,       // [3H]
                     const uint16_t* __restrict__ h_in,  // [B, H] bf16
                     uint16_t* __restrict__ h_out,       // [B, H] bf16
                     uint16_t* __restrict__ Hseq,        // [B*T, H] bf16
                     int t) {
    const int lane = threadIdx.x & 31;
    const int mt = threadIdx.x >> 5;    // wave id = batch row tile: 0..3
    const int jt = blockIdx.x;          // H column tile: 0..63
    const int mrow = mt * 16 + (lane & 15);    // batch index for A fragment
    const int ncol = jt * 16 + (lane & 15);    // H column for B fragment / C tile

    v8f acc_rz0 = {};   // r gate (x + h parts combined)
    v8f acc_rz1 = {};   // z gate (x + h parts combined)
    v8f acc_xn  = {};   // x@Wx_n
    v8f acc_hn  = {};   // h@Wh_n

    // ---- input GEMM: x_t @ Wx  (K = D = 512) ----
    const uint16_t* arow = X + ((size_t)mrow * TT + t) * DD;
    const uint16_t* bxr = WxT + (size_t)(0 * HH + ncol) * DD;
    const uint16_t* bxz = WxT + (size_t)(1 * HH + ncol) * DD;
    const uint16_t* bxn = WxT + (size_t)(2 * HH + ncol) * DD;
    for (int k = 0; k < DD; k += 32) {
        v16bf a = load_a_frag(arow, k);
        acc_rz0 = wmma_bf16(a, load_b_frag(bxr, k), acc_rz0);
        acc_rz1 = wmma_bf16(a, load_b_frag(bxz, k), acc_rz1);
        acc_xn  = wmma_bf16(a, load_b_frag(bxn, k), acc_xn);
    }

    // ---- recurrent GEMM: h @ Wh  (K = H = 1024) ----
    const uint16_t* hrow = h_in + (size_t)mrow * HH;
    const uint16_t* bhr = WhT + (size_t)(0 * HH + ncol) * HH;
    const uint16_t* bhz = WhT + (size_t)(1 * HH + ncol) * HH;
    const uint16_t* bhn = WhT + (size_t)(2 * HH + ncol) * HH;
    for (int k = 0; k < HH; k += 32) {
        v16bf a = load_a_frag(hrow, k);
        acc_rz0 = wmma_bf16(a, load_b_frag(bhr, k), acc_rz0);
        acc_rz1 = wmma_bf16(a, load_b_frag(bhz, k), acc_rz1);
        acc_hn  = wmma_bf16(a, load_b_frag(bhn, k), acc_hn);
    }

    // ---- gate epilogue ----
    const float br_  = bx[ncol]      + bh[ncol];
    const float bz_  = bx[HH + ncol] + bh[HH + ncol];
    const float bxn_ = bx[2 * HH + ncol];
    const float bhn_ = bh[2 * HH + ncol];
    const int rbase = mt * 16 + 8 * (lane >> 4);   // C/D layout: M = vgpr + 8*(lane>=16)
#pragma unroll
    for (int v = 0; v < 8; ++v) {
        const int row = rbase + v;
        float r = sigmoidf_(acc_rz0[v] + br_);
        float z = sigmoidf_(acc_rz1[v] + bz_);
        float n = tanhf(acc_xn[v] + bxn_ + r * (acc_hn[v] + bhn_));
        float hp = bf2f(h_in[(size_t)row * HH + ncol]);
        float hnew = (1.0f - z) * n + z * hp;
        uint16_t hb = f2bf(hnew);
        h_out[(size_t)row * HH + ncol] = hb;
        Hseq[((size_t)row * TT + t) * HH + ncol] = hb;
    }
}

// ---------- output projection: tanh(Hseq @ W_out + b_out) ----------
// grid = (256 M super-tiles, 32 N tiles); block = 4 waves over 4 consecutive
// M tiles sharing the same B columns (WGP$ reuse).
__global__ __launch_bounds__(128)
void out_gemm_kernel(const uint16_t* __restrict__ Hseq,  // [B*T, H] bf16
                     const uint16_t* __restrict__ WoT,   // [D, H] bf16 (transposed)
                     const float* __restrict__ b_out,    // [D]
                     float* __restrict__ out) {          // [B*T, D] f32
    const int lane = threadIdx.x & 31;
    const int mtile = blockIdx.x * 4 + (threadIdx.x >> 5);  // 0..1023
    const int nt = blockIdx.y;                              // 0..31
    const int mrow = mtile * 16 + (lane & 15);
    const int ncol = nt * 16 + (lane & 15);

    const uint16_t* arow = Hseq + (size_t)mrow * HH;
    const uint16_t* bcol = WoT + (size_t)ncol * HH;
    v8f acc = {};
    for (int k = 0; k < HH; k += 32)
        acc = wmma_bf16(load_a_frag(arow, k), load_b_frag(bcol, k), acc);

    const float bo = b_out[ncol];
    const int rbase = mtile * 16 + 8 * (lane >> 4);
#pragma unroll
    for (int v = 0; v < 8; ++v)
        out[(size_t)(rbase + v) * DD + ncol] = tanhf(acc[v] + bo);
}

// ---------- host launcher ----------
extern "C" void kernel_launch(void* const* d_in, const int* in_sizes, int n_in,
                              void* d_out, int out_size, void* d_ws, size_t ws_size,
                              hipStream_t stream) {
    (void)in_sizes; (void)n_in; (void)out_size; (void)ws_size;
    const int*   y      = (const int*)d_in[0];
    const float* hidden = (const float*)d_in[1];
    const float* emb    = (const float*)d_in[2];
    const float* Wx     = (const float*)d_in[3];
    const float* Wh     = (const float*)d_in[4];
    const float* bx     = (const float*)d_in[5];
    const float* bh     = (const float*)d_in[6];
    const float* W_out  = (const float*)d_in[7];
    const float* b_out  = (const float*)d_in[8];
    float* out = (float*)d_out;

    // workspace carve-up (bf16 buffers)
    uint8_t* ws = (uint8_t*)d_ws;
    size_t off = 0;
    auto carve = [&](size_t elems) {
        void* p = ws + off;
        off = (off + elems * sizeof(uint16_t) + 255) & ~(size_t)255;
        return (uint16_t*)p;
    };
    uint16_t* WxT  = carve((size_t)H3 * DD);        // [3H, D]
    uint16_t* WhT  = carve((size_t)H3 * HH);        // [3H, H]
    uint16_t* WoT  = carve((size_t)DD * HH);        // [D, H]
    uint16_t* X    = carve((size_t)BB * TT * DD);   // [B*T, D]
    uint16_t* Hseq = carve((size_t)BB * TT * HH);   // [B*T, H]
    uint16_t* h0   = carve((size_t)BB * HH);        // ping
    uint16_t* h1   = carve((size_t)BB * HH);        // pong

    const int TB = 256;
    auto nb = [&](size_t total) { return (unsigned)((total + TB - 1) / TB); };

    // 1) convert + transpose weights to bf16 (LDS-tiled, coalesced both sides)
    transpose_cvt_kernel<<<dim3(H3 / 32, DD / 32), dim3(32, 8), 0, stream>>>(Wx, WxT, DD, H3);
    transpose_cvt_kernel<<<dim3(H3 / 32, HH / 32), dim3(32, 8), 0, stream>>>(Wh, WhT, HH, H3);
    transpose_cvt_kernel<<<dim3(DD / 32, HH / 32), dim3(32, 8), 0, stream>>>(W_out, WoT, HH, DD);

    // 2) embedding gather -> bf16 X ; initial hidden -> bf16 ping
    gather_cvt_kernel<<<nb((size_t)BB * TT * DD), TB, 0, stream>>>(y, emb, X);
    cvt_f32_bf16_kernel<<<nb((size_t)BB * HH), TB, 0, stream>>>(hidden, h0, BB * HH);

    // 3) sequential GRU recurrence (ping-pong hidden)
    uint16_t* hin = h0;
    uint16_t* hout = h1;
    for (int t = 0; t < TT; ++t) {
        gru_step_kernel<<<HH / 16, 128, 0, stream>>>(
            X, WxT, WhT, bx, bh, hin, hout, Hseq, t);
        uint16_t* tmp = hin; hin = hout; hout = tmp;
    }

    // 4) output projection + tanh
    out_gemm_kernel<<<dim3((BB * TT) / 64, DD / 16), 128, 0, stream>>>(
        Hseq, WoT, b_out, out);
}
